// PairAttention_41162966565150
// MI455X (gfx1250) — compile-verified, matrix operand order checked by reading, not run on hardware
//
#include <hip/hip_runtime.h>
#include <hip/hip_bf16.h>

typedef __attribute__((ext_vector_type(16))) _Float16 v16h;
typedef __attribute__((ext_vector_type(8)))  _Float16 v8h;
typedef __attribute__((ext_vector_type(8)))  float    v8f;

#define NUM_HEADS 8
#define D_MODEL   144
#define HEAD_DIM  18
#define KPAD      160      // 144 padded to multiple of 32 (WMMA K)
#define HDPAD     32       // 18 padded to WMMA K
#define BATCH     2
#define LSEQ      48
#define TOK_PER_B (LSEQ*LSEQ)      // 2304
#define NTOK      (BATCH*TOK_PER_B) // 4608
#define LLEN      TOK_PER_B
#define NCT       (LLEN/16)        // 144 column tiles of scores
#define NKT32     (LLEN/32)        // 72 K-tiles for P*V
#define LN_EPS    1e-5f

// Load a 16-bit WMMA fragment: two contiguous 8-half (16B) chunks at +0,+16.
// Works for both A (lane = row&15) and B (lane = col&15) with the documented
// CDNA5 16-bit layout; caller passes base already offset by lane row/col and
// the lane-half K offset (0 for lanes 0-15, 8 for lanes 16-31).
__device__ inline v16h load_frag16(const _Float16* p) {
  union { v16h v; v8h h[2]; } u;
  u.h[0] = *(const v8h*)(p);
  u.h[1] = *(const v8h*)(p + 16);
  return u.v;
}

// ---------------- LayerNorm -> f16 (K-padded) ----------------
__global__ void ln_kernel(const float* __restrict__ z, const float* __restrict__ w,
                          const float* __restrict__ bb, _Float16* __restrict__ znh) {
  int token = blockIdx.x * (blockDim.x >> 5) + (threadIdx.x >> 5);
  int lane  = threadIdx.x & 31;
  if (token >= NTOK) return;
  const float* zp = z + (size_t)token * D_MODEL;
  float s = 0.f, ss = 0.f;
  for (int i = lane; i < D_MODEL; i += 32) { float v = zp[i]; s += v; ss += v * v; }
  for (int off = 16; off > 0; off >>= 1) {
    s  += __shfl_xor(s,  off, 32);
    ss += __shfl_xor(ss, off, 32);
  }
  float mu   = s * (1.f / D_MODEL);
  float var  = ss * (1.f / D_MODEL) - mu * mu;
  float rstd = rsqrtf(var + LN_EPS);
  _Float16* op = znh + (size_t)token * KPAD;
  for (int i = lane; i < D_MODEL; i += 32)
    op[i] = (_Float16)(((zp[i] - mu) * rstd) * w[i] + bb[i]);
  for (int i = D_MODEL + lane; i < KPAD; i += 32) op[i] = (_Float16)0.f;
}

// ---------------- fp32 weights [N,144] -> f16 [N,160] zero-padded ----------------
__global__ void pad_weights(const float* __restrict__ src, _Float16* __restrict__ dst, int n) {
  int idx = blockIdx.x * blockDim.x + threadIdx.x;
  if (idx >= n * KPAD) return;
  int r = idx / KPAD, c = idx % KPAD;
  dst[idx] = (c < D_MODEL) ? (_Float16)src[r * D_MODEL + c] : (_Float16)0.f;
}

// ---------------- Generic WMMA GEMM: C[M,N] = A[M,KPAD] * W[N,KPAD]^T ----------------
// 4 waves / block, one 16x16 output tile per wave.
__global__ void gemm_wmma(const _Float16* __restrict__ A, const _Float16* __restrict__ W,
                          float* __restrict__ C, int M, int N) {
  int lane = threadIdx.x & 31;
  int wave = threadIdx.x >> 5;
  int mt   = blockIdx.y * 4 + wave;
  int nt   = blockIdx.x;
  if (mt * 16 >= M) return;
  int rsel  = lane & 15;
  int khalf = (lane < 16) ? 0 : 8;
  const _Float16* arow = A + (size_t)(mt * 16 + rsel) * KPAD + khalf;
  const _Float16* wrow = W + (size_t)(nt * 16 + rsel) * KPAD + khalf;
  v8f acc = {};
#pragma unroll
  for (int kt = 0; kt < KPAD / 32; ++kt) {
    v16h a = load_frag16(arow + kt * 32);
    v16h b = load_frag16(wrow + kt * 32);
    acc = __builtin_amdgcn_wmma_f32_16x16x32_f16(false, a, false, b, (short)0, acc, false, false);
  }
  int rbase = (lane >> 4) * 8;
  float* crow = C + (size_t)(mt * 16 + rbase) * N + nt * 16 + rsel;
#pragma unroll
  for (int r = 0; r < 8; ++r) crow[(size_t)r * N] = acc[r];
}

// ---------------- Split QKV: f32 [4608,432] -> per-head f16, Q scaled ----------------
__global__ void qkv_split(const float* __restrict__ qkv, _Float16* __restrict__ qh,
                          _Float16* __restrict__ kh, _Float16* __restrict__ vT) {
  int idx = blockIdx.x * blockDim.x + threadIdx.x;
  const int total = BATCH * NUM_HEADS * TOK_PER_B * HDPAD;
  if (idx >= total) return;
  int d   = idx & (HDPAD - 1);
  int tmp = idx >> 5;
  int p   = tmp % TOK_PER_B;
  int bh  = tmp / TOK_PER_B;
  int h   = bh % NUM_HEADS, b = bh / NUM_HEADS;
  size_t t = (size_t)b * TOK_PER_B + p;
  const float scale = 0.2357022603955158f;  // 1/sqrt(18)
  float qv = 0.f, kv = 0.f, vv = 0.f;
  if (d < HEAD_DIM) {
    const float* row = qkv + t * (3 * D_MODEL);
    qv = row[h * HEAD_DIM + d] * scale;
    kv = row[D_MODEL + h * HEAD_DIM + d];
    vv = row[2 * D_MODEL + h * HEAD_DIM + d];
  }
  qh[idx] = (_Float16)qv;
  kh[idx] = (_Float16)kv;
  vT[((size_t)bh * HDPAD + d) * TOK_PER_B + p] = (_Float16)vv;  // transposed for B-frag
}

// ---------------- Attention: one (head, 16-query block) per workgroup ----------------
// LDS: S f32[16][2304] | P f16[16][2304] | red f32[16*8] | rowstat f32[32] | partial f32[4][16][32]
__global__ void attn_kernel(const _Float16* __restrict__ qh, const _Float16* __restrict__ kh,
                            const _Float16* __restrict__ vT, const float* __restrict__ mask,
                            _Float16* __restrict__ outh) {
  extern __shared__ char smem[];
  float*    S       = (float*)smem;                         // 147456 B
  _Float16* P       = (_Float16*)(smem + 16 * LLEN * 4);    //  73728 B
  float*    red     = (float*)(smem + 16 * LLEN * 6);       //    512 B
  float*    rowstat = red + 128;                            //    128 B
  float*    partial = rowstat + 32;                         //   8192 B

  int bh = blockIdx.x, qtile = blockIdx.y;
  int b = bh / NUM_HEADS, h = bh % NUM_HEADS;
  int lane = threadIdx.x & 31, wave = threadIdx.x >> 5;
  int rsel = lane & 15;
  int khalf = (lane < 16) ? 0 : 8;
  int qbase = qtile * 16;
  const float* mrow = mask + (size_t)b * LLEN;

  // Q fragment (single K-tile: hd padded to 32); scale pre-folded into Q.
  v16h afrag = load_frag16(qh + ((size_t)bh * LLEN + qbase + rsel) * HDPAD + khalf);

  // --- Phase 1: S = (Q Kt) masked, waves stripe the 144 column tiles ---
  for (int ct = wave; ct < NCT; ct += 4) {
    v16h bfrag = load_frag16(kh + ((size_t)bh * LLEN + ct * 16 + rsel) * HDPAD + khalf);
    v8f c = {};
    c = __builtin_amdgcn_wmma_f32_16x16x32_f16(false, afrag, false, bfrag, (short)0, c, false, false);
    int col = ct * 16 + rsel;
    float mk = mrow[col];
    int rowoff = (lane >> 4) * 8;
#pragma unroll
    for (int r = 0; r < 8; ++r) {
      int row = rowoff + r;
      float mq = mrow[qbase + row];
      S[row * LLEN + col] = (mq * mk != 0.f) ? c[r] : -1e9f;
    }
  }
  __syncthreads();

  // --- Phase 2: softmax over each of the 16 rows (2304 wide) ---
  int r = threadIdx.x >> 3;   // 0..15 row
  int j = threadIdx.x & 7;    // 0..7 chunk
  const int CH = LLEN / 8;    // 288
  float* srow = S + r * LLEN + j * CH;
  float lmax = -3.4e38f;
  for (int c0 = 0; c0 < CH; ++c0) lmax = fmaxf(lmax, srow[c0]);
  red[r * 8 + j] = lmax;
  __syncthreads();
  if (j == 0) {
    float m = red[r * 8];
    for (int t = 1; t < 8; ++t) m = fmaxf(m, red[r * 8 + t]);
    rowstat[r] = m;
  }
  __syncthreads();
  float rmax = rowstat[r];
  float lsum = 0.f;
  for (int c0 = 0; c0 < CH; ++c0) { float e = __expf(srow[c0] - rmax); srow[c0] = e; lsum += e; }
  red[r * 8 + j] = lsum;
  __syncthreads();
  if (j == 0) {
    float s2 = 0.f;
    for (int t = 0; t < 8; ++t) s2 += red[r * 8 + t];
    rowstat[16 + r] = s2;
  }
  __syncthreads();
  float inv = 1.0f / rowstat[16 + r];
  _Float16* prow = P + r * LLEN + j * CH;
  for (int c0 = 0; c0 < CH; ++c0) prow[c0] = (_Float16)(srow[c0] * inv);
  __syncthreads();

  // --- Phase 3: O = P * V; waves split the 72 K-tiles, reduce in LDS ---
  v8f acc0 = {}, acc1 = {};
  const _Float16* vbase = vT + (size_t)bh * HDPAD * LLEN;
  for (int kt = wave * (NKT32 / 4); kt < (wave + 1) * (NKT32 / 4); ++kt) {
    v16h a  = load_frag16(P + rsel * LLEN + kt * 32 + khalf);
    v16h b0 = load_frag16(vbase + (size_t)rsel * LLEN + kt * 32 + khalf);
    v16h b1 = load_frag16(vbase + (size_t)(16 + rsel) * LLEN + kt * 32 + khalf);
    acc0 = __builtin_amdgcn_wmma_f32_16x16x32_f16(false, a, false, b0, (short)0, acc0, false, false);
    acc1 = __builtin_amdgcn_wmma_f32_16x16x32_f16(false, a, false, b1, (short)0, acc1, false, false);
  }
  {
    int rowoff = (lane >> 4) * 8;
    float* pw = partial + wave * 16 * 32;
#pragma unroll
    for (int rr = 0; rr < 8; ++rr) {
      pw[(rowoff + rr) * 32 + rsel]      = acc0[rr];
      pw[(rowoff + rr) * 32 + 16 + rsel] = acc1[rr];
    }
  }
  __syncthreads();
  for (int idx = threadIdx.x; idx < 16 * 32; idx += blockDim.x) {
    int rr = idx >> 5, cc = idx & 31;
    float s = partial[idx] + partial[512 + idx] + partial[1024 + idx] + partial[1536 + idx];
    if (cc < HEAD_DIM)
      outh[((size_t)b * TOK_PER_B + qbase + rr) * KPAD + h * HEAD_DIM + cc] = (_Float16)s;
  }
  if (h == 0) {  // zero the K-pad columns of the attention-output A matrix
    for (int idx = threadIdx.x; idx < 16 * (KPAD - D_MODEL); idx += blockDim.x) {
      int rr = idx / (KPAD - D_MODEL), cc = idx % (KPAD - D_MODEL);
      outh[((size_t)b * TOK_PER_B + qbase + rr) * KPAD + D_MODEL + cc] = (_Float16)0.f;
    }
  }
}

// ---------------- finalize: out = proj * sigmoid(gate) * mask ----------------
__global__ void finalize_kernel(const float* __restrict__ proj, const float* __restrict__ gpre,
                                const float* __restrict__ mask, float* __restrict__ out) {
  int idx = blockIdx.x * blockDim.x + threadIdx.x;
  if (idx >= NTOK * D_MODEL) return;
  int t = idx / D_MODEL;
  float g = 1.0f / (1.0f + __expf(-gpre[idx]));
  out[idx] = proj[idx] * g * mask[t];
}

extern "C" void kernel_launch(void* const* d_in, const int* in_sizes, int n_in,
                              void* d_out, int out_size, void* d_ws, size_t ws_size,
                              hipStream_t stream) {
  const float* z      = (const float*)d_in[0];
  const float* pmask  = (const float*)d_in[1];
  const float* norm_w = (const float*)d_in[2];
  const float* norm_b = (const float*)d_in[3];
  const float* w_qkv  = (const float*)d_in[4];
  const float* w_g    = (const float*)d_in[5];
  const float* w_out  = (const float*)d_in[6];
  float* out = (float*)d_out;

  // Workspace carve-up (256B aligned), total ~23.5 MB.
  size_t off = 0;
  auto carve = [&](size_t bytes) {
    void* p = (char*)d_ws + off;
    off = (off + bytes + 255) & ~(size_t)255;
    return p;
  };
  _Float16* znh    = (_Float16*)carve((size_t)NTOK * KPAD * 2);
  _Float16* wqkvh  = (_Float16*)carve((size_t)3 * D_MODEL * KPAD * 2);
  _Float16* wgh    = (_Float16*)carve((size_t)D_MODEL * KPAD * 2);
  _Float16* wouth  = (_Float16*)carve((size_t)D_MODEL * KPAD * 2);
  float*    qkv    = (float*)   carve((size_t)NTOK * 3 * D_MODEL * 4);
  _Float16* qh     = (_Float16*)carve((size_t)BATCH * NUM_HEADS * LLEN * HDPAD * 2);
  _Float16* kh     = (_Float16*)carve((size_t)BATCH * NUM_HEADS * LLEN * HDPAD * 2);
  _Float16* vT     = (_Float16*)carve((size_t)BATCH * NUM_HEADS * LLEN * HDPAD * 2);
  _Float16* outh   = (_Float16*)carve((size_t)NTOK * KPAD * 2);
  float*    proj   = (float*)   carve((size_t)NTOK * D_MODEL * 4);
  float*    gpre   = (float*)   carve((size_t)NTOK * D_MODEL * 4);
  (void)ws_size; (void)in_sizes; (void)n_in; (void)out_size;

  // 1) LayerNorm -> f16 padded
  ln_kernel<<<NTOK / 8, 256, 0, stream>>>(z, norm_w, norm_b, znh);

  // 2) Pad/convert weights
  pad_weights<<<(3 * D_MODEL * KPAD + 255) / 256, 256, 0, stream>>>(w_qkv, wqkvh, 3 * D_MODEL);
  pad_weights<<<(D_MODEL * KPAD + 255) / 256, 256, 0, stream>>>(w_g,   wgh,   D_MODEL);
  pad_weights<<<(D_MODEL * KPAD + 255) / 256, 256, 0, stream>>>(w_out, wouth, D_MODEL);

  // 3) QKV projection: [4608,160] x [432,160]^T -> [4608,432]
  gemm_wmma<<<dim3((3 * D_MODEL) / 16, NTOK / 64), 128, 0, stream>>>(znh, wqkvh, qkv, NTOK, 3 * D_MODEL);

  // 4) Split + scale + transpose V
  {
    int total = BATCH * NUM_HEADS * TOK_PER_B * HDPAD;
    qkv_split<<<(total + 255) / 256, 256, 0, stream>>>(qkv, qh, kh, vT);
  }

  // 5) Attention (16 heads x 144 query tiles), 230016 B dynamic LDS
  attn_kernel<<<dim3(BATCH * NUM_HEADS, LLEN / 16), 128, 16 * LLEN * 6 + 512 + 128 + 8192, stream>>>(
      qh, kh, vT, pmask, outh);

  // 6) Gate pre-activation and output projection
  gemm_wmma<<<dim3(D_MODEL / 16, NTOK / 64), 128, 0, stream>>>(znh,  wgh,   gpre, NTOK, D_MODEL);
  gemm_wmma<<<dim3(D_MODEL / 16, NTOK / 64), 128, 0, stream>>>(outh, wouth, proj, NTOK, D_MODEL);

  // 7) Gate * mask
  finalize_kernel<<<(NTOK * D_MODEL + 255) / 256, 256, 0, stream>>>(proj, gpre, pmask, out);
}
